// Attention_27582279975411
// MI455X (gfx1250) — matrix operand order repaired and checked by reading, not compiled, not run on hardware
//
#include <hip/hip_runtime.h>

// ---------------------------------------------------------------------------
// MI455X (gfx1250) causal self-attention, bf16 WMMA pipeline.
// Shapes fixed by reference: B=2, S=2048, D=2048, H=16, Dh=128.
// All heavy math on v_wmma_f32_16x16x32_bf16 (f32 accumulate).
// ---------------------------------------------------------------------------

typedef __bf16 bf16;
typedef __attribute__((ext_vector_type(16))) __bf16 v16bf;
typedef __attribute__((ext_vector_type(8)))  float  v8f;

#define LN_ROPE_BASE 9.210340371976184f   /* ln(10000) */
#define ATTN_SCALE   0.08838834764831845f /* 1/sqrt(128), folded into Q */

// --- WMMA fragment K-index maps (CDNA5 ISA §7.12.2, 16-bit) -----
// A-matrix 16x32 (contiguous order chosen here, not the ISA table's): lanes 0-15 hold K 0-15, lanes 16-31 hold K 16-31.
static __device__ __forceinline__ int kmapA(int j, int half) {
    return j + 16 * half;
}
// B-matrix 32x16 (same contiguous order as A, not the ISA table's): lanes 0-15 hold K 0-15, lanes 16-31 hold K 16-31.
static __device__ __forceinline__ int kmapB(int j, int half) {
    return j + 16 * half;
}

// A fragment: rows m = row0 + lane%16 of row-major [.,ld] bf16 matrix.
static __device__ __forceinline__ v16bf load_a_frag(const bf16* __restrict__ p,
                                                    int ld, int row0, int k0) {
    const int lane = threadIdx.x & 31;
    const int half = lane >> 4;
    const bf16* base = p + (size_t)(row0 + (lane & 15)) * ld + k0;
    v16bf f;
#pragma unroll
    for (int j = 0; j < 16; ++j) f[j] = base[kmapA(j, half)];
    return f;
}

// B fragment formed from a row-major [N,K] matrix used transposed:
// B[k][n] = W[n0+lane%16][k0+kmapB].
static __device__ __forceinline__ v16bf load_bt_frag(const bf16* __restrict__ p,
                                                     int ld, int n0, int k0) {
    const int lane = threadIdx.x & 31;
    const int half = lane >> 4;
    const bf16* base = p + (size_t)(n0 + (lane & 15)) * ld + k0;
    v16bf f;
#pragma unroll
    for (int j = 0; j < 16; ++j) f[j] = base[kmapB(j, half)];
    return f;
}

// B fragment from a row-major [K,N] matrix (V in PV): B[k][n] = M[k0+kmapB][n0+lane%16].
static __device__ __forceinline__ v16bf load_b_frag(const bf16* __restrict__ p,
                                                    int ld, int k0, int n0) {
    const int lane = threadIdx.x & 31;
    const int half = lane >> 4;
    const int n    = n0 + (lane & 15);
    v16bf f;
#pragma unroll
    for (int j = 0; j < 16; ++j) f[j] = p[(size_t)(k0 + kmapB(j, half)) * ld + n];
    return f;
}

// --- elementwise f32 -> bf16 -----------------------------------------------
__global__ void cvt_f32_bf16_kernel(const float* __restrict__ in,
                                    bf16* __restrict__ out, size_t n) {
    size_t i = (size_t)blockIdx.x * blockDim.x + threadIdx.x;
    if (i < n) out[i] = (bf16)in[i];
}

// --- generic bf16 WMMA GEMM: C[M,N] = A[M,K] * W[N,K]^T ---------------------
// 4 waves / block, each wave computes a 64x64 tile of a 128x128 block tile.
template <int N, bool F32OUT>
__global__ __launch_bounds__(128) void gemm_bf16_wmma_kernel(
    const bf16* __restrict__ A, const bf16* __restrict__ W,
    float* __restrict__ Cf, bf16* __restrict__ Cb, int Kd) {
    const int lane = threadIdx.x & 31;
    const int wave = threadIdx.x >> 5;
    const int half = lane >> 4;
    const int ln   = lane & 15;
    const int row0 = blockIdx.x * 128 + (wave >> 1) * 64;
    const int col0 = blockIdx.y * 128 + (wave & 1) * 64;

    v8f acc[4][4] = {};
    for (int k0 = 0; k0 < Kd; k0 += 32) {
        v16bf a[4], b[4];
#pragma unroll
        for (int i = 0; i < 4; ++i) a[i] = load_a_frag(A, Kd, row0 + 16 * i, k0);
#pragma unroll
        for (int j = 0; j < 4; ++j) b[j] = load_bt_frag(W, Kd, col0 + 16 * j, k0);
        // pull the tiles 8 K-steps ahead toward the WGP (global_prefetch_b8)
        if (k0 + 256 < Kd) {
#pragma unroll
            for (int i = 0; i < 4; ++i) {
                __builtin_prefetch(A + (size_t)(row0 + 16 * i + (lane & 15)) * Kd + k0 + 256, 0, 3);
                __builtin_prefetch(W + (size_t)(col0 + 16 * i + (lane & 15)) * Kd + k0 + 256, 0, 3);
            }
        }
#pragma unroll
        for (int i = 0; i < 4; ++i)
#pragma unroll
            for (int j = 0; j < 4; ++j)
                acc[i][j] = __builtin_amdgcn_wmma_f32_16x16x32_bf16(
                    false, a[i], false, b[j], (short)0, acc[i][j], false, false);
    }
    // C 16x16 f32 layout: col = lane%16, row = e + 8*(lane/16).
    const size_t base = (size_t)(row0 + 8 * half) * N + (size_t)(col0 + ln);
    if (F32OUT) {
        float* __restrict__ p = Cf + base;
#pragma unroll
        for (int i = 0; i < 4; ++i)
#pragma unroll
            for (int e = 0; e < 8; ++e)
#pragma unroll
                for (int j = 0; j < 4; ++j)
                    p[(size_t)(16 * i + e) * N + 16 * j] = acc[i][j][e];
    } else {
        bf16* __restrict__ p = Cb + base;
#pragma unroll
        for (int i = 0; i < 4; ++i)
#pragma unroll
            for (int e = 0; e < 8; ++e)
#pragma unroll
                for (int j = 0; j < 4; ++j)
                    p[(size_t)(16 * i + e) * N + 16 * j] = (bf16)acc[i][j][e];
    }
}

// --- RoPE + split qkv [B,S,3D] bf16 -> Q,K,V [B,H,S,Dh] bf16 ----------------
// Q is pre-scaled by 1/sqrt(Dh) so the attention kernel needs no score scaling.
__global__ void rope_split_kernel(const bf16* __restrict__ qkv,
                                  bf16* __restrict__ Q, bf16* __restrict__ K,
                                  bf16* __restrict__ V, int B, int H, int S) {
    const int Dh = 128, D = 2048;
    size_t idx   = (size_t)blockIdx.x * blockDim.x + threadIdx.x;
    size_t total = (size_t)B * H * S * 64;
    if (idx >= total) return;
    const int i  = (int)(idx & 63);
    const int s  = (int)((idx >> 6) % S);
    const int hh = (int)(((idx >> 6) / S) % H);
    const int b  = (int)(idx / ((size_t)64 * S * H));

    const float inv = __expf(-((float)i / 64.f) * LN_ROPE_BASE);
    const float f   = (float)s * inv;
    const float c   = __cosf(f);
    const float sn  = __sinf(f);

    const bf16* src = qkv + (size_t)(b * S + s) * (3 * D) + hh * Dh;
    const size_t dst = ((size_t)(b * H + hh) * S + s) * Dh;

    {   // Q with RoPE, pre-scaled
        float x1 = (float)src[i], x2 = (float)src[i + 64];
        Q[dst + i]      = (bf16)((x1 * c - x2 * sn) * ATTN_SCALE);
        Q[dst + i + 64] = (bf16)((x2 * c + x1 * sn) * ATTN_SCALE);
    }
    {   // K with RoPE
        const bf16* sk = src + D;
        float x1 = (float)sk[i], x2 = (float)sk[i + 64];
        K[dst + i]      = (bf16)(x1 * c - x2 * sn);
        K[dst + i + 64] = (bf16)(x2 * c + x1 * sn);
    }
    {   // V passthrough
        const bf16* sv = src + 2 * D;
        V[dst + i]      = sv[i];
        V[dst + i + 64] = sv[i + 64];
    }
}

// --- flash attention (causal, online softmax) -------------------------------
// grid (S/64, B*H), 4 waves/block; each wave owns 16 q-rows. Uniform kv loop
// so __syncthreads is non-divergent; masking zeroes the overrun.
// Per 32-kv step: both 16x16 score tiles computed first, ONE row-max over all
// 32 columns, one exp/alpha rescale (correct online softmax), P staged via
// per-wave LDS into A-layout, row sums via a ones-matrix WMMA.
__global__ __launch_bounds__(128) void flash_attn_kernel(
    const bf16* __restrict__ Q, const bf16* __restrict__ K,
    const bf16* __restrict__ V, bf16* __restrict__ Z,
    int B, int H, int S) {
    const int Dh = 128, D = 2048;
    __shared__ bf16 Plds[4][16 * 32];

    const int lane = threadIdx.x & 31;
    const int wave = threadIdx.x >> 5;
    const int half = lane >> 4;
    const int ln   = lane & 15;
    const int qb   = blockIdx.x;
    const int bh   = blockIdx.y;
    const int b    = bh / H;
    const int h    = bh % H;

    const bf16* Qh = Q + (size_t)bh * S * Dh;
    const bf16* Kh = K + (size_t)bh * S * Dh;
    const bf16* Vh = V + (size_t)bh * S * Dh;
    const int q0   = qb * 64 + wave * 16;

    v16bf qf[4];
#pragma unroll
    for (int t = 0; t < 4; ++t) qf[t] = load_a_frag(Qh, Dh, q0, 32 * t);

    v16bf ones;
#pragma unroll
    for (int j = 0; j < 16; ++j) ones[j] = (bf16)1.0f;

    v8f acc[8] = {};
    v8f accl = {};          // running row sums (every lane: sum for row e+8*half)
    float mrow[8];
#pragma unroll
    for (int e = 0; e < 8; ++e) mrow[e] = -1e30f;

    const int kv_end = qb * 64 + 64;  // uniform across the 4 waves
    for (int j0 = 0; j0 < kv_end; j0 += 32) {
        // ---- scores for 32 kv columns: two 16x16 tiles, K frags preloaded ----
        v16bf kf0[4], kf1[4];
#pragma unroll
        for (int t = 0; t < 4; ++t) kf0[t] = load_bt_frag(Kh, Dh, j0, 32 * t);
#pragma unroll
        for (int t = 0; t < 4; ++t) kf1[t] = load_bt_frag(Kh, Dh, j0 + 16, 32 * t);
        v8f sc0 = {}, sc1 = {};
#pragma unroll
        for (int t = 0; t < 4; ++t) {
            sc0 = __builtin_amdgcn_wmma_f32_16x16x32_bf16(
                false, qf[t], false, kf0[t], (short)0, sc0, false, false);
            sc1 = __builtin_amdgcn_wmma_f32_16x16x32_bf16(
                false, qf[t], false, kf1[t], (short)0, sc1, false, false);
        }
        // ---- causal mask (finite sentinel, no NaN paths) ----
        const int col0 = j0 + ln, col1 = j0 + 16 + ln;
#pragma unroll
        for (int e = 0; e < 8; ++e) {
            const int row = q0 + e + 8 * half;
            if (col0 > row) sc0[e] = -1e30f;
            if (col1 > row) sc1[e] = -1e30f;
        }
        // ---- one row-max over all 32 columns, stage-wise so the 8 permute
        //      chains pipeline (partial dscnt waits instead of full) ----
        float v[8];
#pragma unroll
        for (int e = 0; e < 8; ++e) v[e] = fmaxf(sc0[e], sc1[e]);
#pragma unroll
        for (int m = 1; m < 16; m <<= 1) {
            float t[8];
#pragma unroll
            for (int e = 0; e < 8; ++e) t[e] = __shfl_xor(v[e], m, 32);
#pragma unroll
            for (int e = 0; e < 8; ++e) v[e] = fmaxf(v[e], t[e]);
        }
        // ---- online softmax update + stage P (16x32 bf16, A-layout source) ----
#pragma unroll
        for (int e = 0; e < 8; ++e) {
            const float mnew  = fmaxf(mrow[e], v[e]);
            const float alpha = __expf(mrow[e] - mnew);
            mrow[e] = mnew;
            const float p0 = __expf(sc0[e] - mnew);
            const float p1 = __expf(sc1[e] - mnew);
            accl[e] = accl[e] * alpha;
#pragma unroll
            for (int t = 0; t < 8; ++t) acc[t][e] = acc[t][e] * alpha;
            Plds[wave][(e + 8 * half) * 32 + ln]      = (bf16)p0;
            Plds[wave][(e + 8 * half) * 32 + 16 + ln] = (bf16)p1;
        }
        __syncthreads();
        // ---- P(16x32) x V(32x128) + row sums via ones-matrix WMMA ----
        v16bf pf;
#pragma unroll
        for (int j = 0; j < 16; ++j)
            pf[j] = Plds[wave][ln * 32 + kmapA(j, half)];
        accl = __builtin_amdgcn_wmma_f32_16x16x32_bf16(
            false, pf, false, ones, (short)0, accl, false, false);
        v16bf vf[4];
#pragma unroll
        for (int t = 0; t < 4; ++t) vf[t] = load_b_frag(Vh, Dh, j0, 16 * t);
#pragma unroll
        for (int t = 0; t < 4; ++t)
            acc[t] = __builtin_amdgcn_wmma_f32_16x16x32_bf16(
                false, pf, false, vf[t], (short)0, acc[t], false, false);
#pragma unroll
        for (int t = 0; t < 4; ++t) vf[t] = load_b_frag(Vh, Dh, j0, 64 + 16 * t);
#pragma unroll
        for (int t = 0; t < 4; ++t)
            acc[4 + t] = __builtin_amdgcn_wmma_f32_16x16x32_bf16(
                false, pf, false, vf[t], (short)0, acc[4 + t], false, false);
        __syncthreads();
    }

    // normalize and scatter to Z [B,S,D] bf16 (mul-only epilogue)
    float rinv[8];
#pragma unroll
    for (int e = 0; e < 8; ++e) rinv[e] = 1.f / accl[e];
    bf16* Zr = Z + ((size_t)b * S + q0 + 8 * half) * D + h * Dh + ln;
#pragma unroll
    for (int e = 0; e < 8; ++e)
#pragma unroll
        for (int t = 0; t < 8; ++t)
            Zr[(size_t)e * D + 16 * t] = (bf16)(acc[t][e] * rinv[e]);
}

// ---------------------------------------------------------------------------
extern "C" void kernel_launch(void* const* d_in, const int* in_sizes, int n_in,
                              void* d_out, int out_size, void* d_ws, size_t ws_size,
                              hipStream_t stream) {
    (void)in_sizes; (void)n_in; (void)out_size; (void)ws_size;
    const int B = 2, S = 2048, D = 2048, H = 16;
    const size_t nX    = (size_t)B * S * D;        // 8388608
    const size_t nWqkv = (size_t)3 * D * D;        // 12582912
    const size_t nWo   = (size_t)D * D;            // 4194304
    const size_t nQKV  = (size_t)B * S * 3 * D;    // 25165824
    const size_t nHead = (size_t)B * H * S * 128;  // 8388608

    const float* x     = (const float*)d_in[0];
    const float* w_qkv = (const float*)d_in[1];
    const float* w_o   = (const float*)d_in[2];
    float*       out   = (float*)d_out;

    // workspace layout (bf16 buffers)
    char* ws = (char*)d_ws;
    bf16* xb    = (bf16*)(ws);                                    // 16 MB
    bf16* wqkvb = (bf16*)(ws + 2 * nX);                           // 25 MB
    bf16* wob   = (bf16*)(ws + 2 * (nX + nWqkv));                 //  8 MB
    bf16* qkvb  = (bf16*)(ws + 2 * (nX + nWqkv + nWo));           // 50 MB
    bf16* Qb    = (bf16*)(ws + 2 * (nX + nWqkv + nWo + nQKV));
    bf16* Kb    = (bf16*)((char*)Qb + 2 * nHead);
    bf16* Vb    = (bf16*)((char*)Kb + 2 * nHead);
    bf16* Zb    = (bf16*)((char*)Vb + 2 * nHead);

    // 1-3: precision conversion (bandwidth-trivial; L2 holds everything after)
    cvt_f32_bf16_kernel<<<(unsigned)((nX    + 255) / 256), 256, 0, stream>>>(x,     xb,    nX);
    cvt_f32_bf16_kernel<<<(unsigned)((nWqkv + 255) / 256), 256, 0, stream>>>(w_qkv, wqkvb, nWqkv);
    cvt_f32_bf16_kernel<<<(unsigned)((nWo   + 255) / 256), 256, 0, stream>>>(w_o,   wob,   nWo);

    // 4: QKV projection  [4096 x 6144] = xb [4096 x 2048] * wqkvb^T
    gemm_bf16_wmma_kernel<6144, false><<<dim3(B * S / 128, 3 * D / 128), 128, 0, stream>>>(
        xb, wqkvb, nullptr, qkvb, D);

    // 5: RoPE + head split (Q pre-scaled by 1/sqrt(Dh))
    {
        size_t total = (size_t)B * H * S * 64;
        rope_split_kernel<<<(unsigned)((total + 255) / 256), 256, 0, stream>>>(
            qkvb, Qb, Kb, Vb, B, H, S);
    }

    // 6: causal flash attention -> Zb [B,S,D] bf16
    flash_attn_kernel<<<dim3(S / 64, B * H), 128, 0, stream>>>(
        Qb, Kb, Vb, Zb, B, H, S);

    // 7: output projection  out [4096 x 2048] f32 = Zb * wob^T
    gemm_bf16_wmma_kernel<2048, true><<<dim3(B * S / 128, D / 128), 128, 0, stream>>>(
        Zb, wob, out, nullptr, D);
}
